// DownsampleONN2d_2259152798478
// MI455X (gfx1250) — compile-verified, matrix operand order checked by reading, not run on hardware
//
#include <hip/hip_runtime.h>

typedef __attribute__((ext_vector_type(16))) _Float16 v16h;
typedef __attribute__((ext_vector_type(8)))  float    v8f;

#define Bn   8
#define Cc   128
#define Hh   256
#define Ww   256
#define Qq   3
#define KC   (Qq*Cc)      /* 384 */
#define OCn  64
#define EPSf 1e-5f
#define HWn  (Hh*Ww)

// ---------------------------------------------------------------------------
// Kernel 0: pack conv weights into WMMA B-fragment layout (f16).
// Wp[tap(9)][kb(12)][ob(4)][lane(32)][16 halves]
// lane<16: col N=lane, K = kb*32 + {0..7, 16..23}
// lane>=16: col N=lane-16, K = kb*32 + {8..15, 24..31}
// ---------------------------------------------------------------------------
__global__ __launch_bounds__(32) void pack_w_kernel(const float* __restrict__ Wc,
                                                    _Float16* __restrict__ Wp) {
    int frag = blockIdx.x;            // 0..431 = (tap*12 + kb)*4 + ob
    int lane = threadIdx.x;           // 0..31
    int ob   = frag & 3;
    int kb   = (frag >> 2) % 12;
    int tap  = frag / 48;
    int kh = tap / 3, kw = tap % 3;
    int n     = ob * 16 + (lane & 15);
    int khalf = lane >> 4;

    union { _Float16 h[16]; uint4 u[2]; } f;
    for (int j = 0; j < 8; ++j) {
        int k1 = kb * 32 + khalf * 8 + j;
        int k2 = k1 + 16;
        f.h[j]     = (_Float16)Wc[(((size_t)n * KC + k1) * 3 + kh) * 3 + kw];
        f.h[8 + j] = (_Float16)Wc[(((size_t)n * KC + k2) * 3 + kh) * 3 + kw];
    }
    uint4* dst = (uint4*)(Wp + (size_t)frag * 512 + (size_t)lane * 16);
    dst[0] = f.u[0];
    dst[1] = f.u[1];
}

// ---------------------------------------------------------------------------
// Kernel 1: per-pixel LayerNorm over channels + powers x^1..x^3, fp32 -> f16,
// written pixel-major/channel-inner: xp[(b*H + h)*W + w][q*128 + c]
// Block: 256 threads, 32 pixels (one row segment).
// ---------------------------------------------------------------------------
__global__ __launch_bounds__(256) void ln_pow_kernel(const float* __restrict__ x,
                                                     const float* __restrict__ g,
                                                     const float* __restrict__ be,
                                                     _Float16* __restrict__ xp) {
    __shared__ float sh[Cc][33];      // 33 pad to dodge bank conflicts in phase C
    __shared__ float ps[8][32];
    __shared__ float ps2[8][32];
    __shared__ float smu[32];
    __shared__ float srs[32];

    int t = threadIdx.x;
    long long pix0 = (long long)blockIdx.x * 32;       // W%32==0: one row, one b
    int bb  = (int)(pix0 / HWn);
    int hw0 = (int)(pix0 % HWn);

    int pix = t & 31;
    int cg8 = t >> 5;                 // 0..7
    const float* xb = x + (size_t)bb * Cc * HWn + hw0;

    // coalesced load: 8 channels x 32 pixels per iteration
    for (int cc = 0; cc < 16; ++cc) {
        int c = cc * 8 + cg8;
        sh[c][pix] = xb[(size_t)c * HWn + pix];
    }
    __syncthreads();

    // partial stats: (pix, cg8) sums 16 channels
    float s = 0.f, s2 = 0.f;
    for (int j = 0; j < 16; ++j) {
        float v = sh[cg8 * 16 + j][pix];
        s += v; s2 += v * v;
    }
    ps[cg8][pix] = s; ps2[cg8][pix] = s2;
    __syncthreads();

    if (t < 32) {
        float a = 0.f, a2 = 0.f;
        for (int gg = 0; gg < 8; ++gg) { a += ps[gg][t]; a2 += ps2[gg][t]; }
        float mu  = a * (1.f / Cc);
        float var = a2 * (1.f / Cc) - mu * mu;
        smu[t] = mu;
        srs[t] = rsqrtf(var + EPSf);
    }
    __syncthreads();

    // write phase: thread -> (pixel = t>>3, 16-channel group = t&7), channel-inner
    int pixw  = t >> 3;
    int cbase = (t & 7) * 16;
    float mu = smu[pixw], rs = srs[pixw];

    union { _Float16 h[16]; uint4 u[2]; } h1, h2, h3;
    for (int j = 0; j < 16; ++j) {
        int c = cbase + j;
        float xn = (sh[c][pixw] - mu) * rs * g[c] + be[c];
        float x2 = xn * xn;
        h1.h[j] = (_Float16)xn;
        h2.h[j] = (_Float16)x2;
        h3.h[j] = (_Float16)(x2 * xn);
    }
    _Float16* dst = xp + ((size_t)pix0 + pixw) * KC;
    uint4* d1 = (uint4*)(dst + cbase);
    uint4* d2 = (uint4*)(dst + Cc + cbase);
    uint4* d3 = (uint4*)(dst + 2 * Cc + cbase);
    d1[0] = h1.u[0]; d1[1] = h1.u[1];
    d2[0] = h2.u[0]; d2[1] = h2.u[1];
    d3[0] = h3.u[0]; d3[1] = h3.u[1];
}

// ---------------------------------------------------------------------------
// Kernel 2: implicit-GEMM 3x3 conv via v_wmma_f32_16x16x32_f16 + fused
// PixelUnshuffle(2) on store.
// Block: 256 thr = 8 waves; each wave: 16-pixel tile x ALL 64 OC
// (4 independent v8f accumulators -> A-fragment reused 4x, 4-way WMMA ILP).
// Per wave: 9 taps x 12 kblocks x 4 OC-blocks = 432 WMMAs.
// ---------------------------------------------------------------------------
__global__ __launch_bounds__(256) void conv_wmma_kernel(const _Float16* __restrict__ xp,
                                                        const _Float16* __restrict__ Wp,
                                                        float* __restrict__ out) {
    int lane = threadIdx.x & 31;
    int wave = threadIdx.x >> 5;      // 0..7 -> pixel tile within block

    int blk = blockIdx.x;             // ((b*256 + h)*2 + wseg)
    int wseg = blk & 1;
    int h0   = (blk >> 1) & 255;
    int b    = blk >> 9;
    int w0   = wseg * 128 + wave * 16;

    int m     = lane & 15;            // A-matrix row = pixel within tile
    int khalf = lane >> 4;

    v8f acc0 = {}, acc1 = {}, acc2 = {}, acc3 = {};

    const uint4 z4 = make_uint4(0u, 0u, 0u, 0u);

    for (int kh = 0; kh < 3; ++kh) {
        int hh = h0 + kh - 1;
        bool hok = (unsigned)hh < (unsigned)Hh;
        for (int kw = 0; kw < 3; ++kw) {
            int ww = w0 + m + kw - 1;
            bool ok = hok && ((unsigned)ww < (unsigned)Ww);
            const _Float16* arow =
                xp + (((size_t)b * Hh + (hok ? hh : 0)) * Ww + (ok ? ww : 0)) * KC;
            int tap = kh * 3 + kw;
            // fragments for the 4 OC-blocks at (tap, kb) are contiguous (+ob*512)
            const _Float16* bbase =
                Wp + ((size_t)tap * 48) * 512 + (size_t)lane * 16;

            for (int kb = 0; kb < 12; ++kb) {
                union { v16h v; uint4 u[2]; } af;
                if (ok) {
                    const uint4* p = (const uint4*)(arow + kb * 32 + khalf * 8);
                    af.u[0] = p[0];       // K = kb*32 + khalf*8 + {0..7}
                    af.u[1] = p[2];       // K = kb*32 + 16 + khalf*8 + {0..7}
                } else {
                    af.u[0] = z4; af.u[1] = z4;
                }
                const uint4* q = (const uint4*)(bbase + (size_t)kb * 4 * 512);
                union { v16h v; uint4 u[2]; } w0f, w1f, w2f, w3f;
                w0f.u[0] = q[0];   w0f.u[1] = q[1];     // ob=0
                w1f.u[0] = q[64];  w1f.u[1] = q[65];    // ob=1 (+512 halves)
                w2f.u[0] = q[128]; w2f.u[1] = q[129];   // ob=2
                w3f.u[0] = q[192]; w3f.u[1] = q[193];   // ob=3

                acc0 = __builtin_amdgcn_wmma_f32_16x16x32_f16(
                    false, af.v, false, w0f.v, (short)0, acc0, false, false);
                acc1 = __builtin_amdgcn_wmma_f32_16x16x32_f16(
                    false, af.v, false, w1f.v, (short)0, acc1, false, false);
                acc2 = __builtin_amdgcn_wmma_f32_16x16x32_f16(
                    false, af.v, false, w2f.v, (short)0, acc2, false, false);
                acc3 = __builtin_amdgcn_wmma_f32_16x16x32_f16(
                    false, af.v, false, w3f.v, (short)0, acc3, false, false);
            }
        }
    }

    // C layout: VGPR i -> (M = i or 8+i, N = lane%16). Scatter with unshuffle:
    // out[b][oc*4 + (h&1)*2 + (w&1)][h/2][w/2]
    int mbase = (lane < 16) ? 0 : 8;
    int h2 = h0 >> 1;
    int s1 = h0 & 1;
    v8f accs[4] = {acc0, acc1, acc2, acc3};
    for (int ob = 0; ob < 4; ++ob) {
        int oc = ob * 16 + (lane & 15);
        for (int i = 0; i < 8; ++i) {
            int w   = w0 + mbase + i;
            int och = oc * 4 + s1 * 2 + (w & 1);
            size_t o = (((size_t)b * 256 + och) * 128 + h2) * 128 + (w >> 1);
            out[o] = accs[ob][i];
        }
    }
}

// ---------------------------------------------------------------------------
extern "C" void kernel_launch(void* const* d_in, const int* in_sizes, int n_in,
                              void* d_out, int out_size, void* d_ws, size_t ws_size,
                              hipStream_t stream) {
    const float* x     = (const float*)d_in[0];
    const float* gamma = (const float*)d_in[1];
    const float* beta  = (const float*)d_in[2];
    const float* Wc    = (const float*)d_in[3];
    float* out = (float*)d_out;

    // workspace layout: Wp (432 frags * 512 halves * 2B = 442368 B, 256-aligned),
    // then xp (8*256*256*384 halves = 402,653,184 B)
    _Float16* Wp = (_Float16*)d_ws;
    _Float16* xp = (_Float16*)((char*)d_ws + 442368);

    pack_w_kernel<<<432, 32, 0, stream>>>(Wc, Wp);
    ln_pow_kernel<<<(Bn * HWn) / 32, 256, 0, stream>>>(x, gamma, beta, xp);
    conv_wmma_kernel<<<Bn * Hh * (Ww / 128), 256, 0, stream>>>(xp, Wp, out);
}
